// CausalSelfAttention_70832600646065
// MI455X (gfx1250) — matrix operand order repaired and checked by reading, not compiled, not run on hardware
//
#include <hip/hip_runtime.h>
#include <hip/hip_bf16.h>

// ---------------------------------------------------------------------------
// CausalSelfAttention (GQA, sliding-window, RoPE, RMSNorm, gated ve) for
// MI455X / gfx1250.  All matmuls via v_wmma_f32_16x16x32_bf16 (wave32 WMMA).
// GEMMs stage operands in LDS with GLOBAL_LOAD_ASYNC_TO_LDS_B128 (ASYNCcnt).
// B=2 T=2048 C=1024 NH=16 NKV=4 HD=64 VEC=32, window read from device scalar.
// ---------------------------------------------------------------------------

constexpr int B_   = 2;
constexpr int T_   = 2048;
constexpr int C_   = 1024;
constexpr int NH_  = 16;
constexpr int NKV_ = 4;
constexpr int HD_  = 64;
constexpr int VEC_ = 32;
constexpr int NQKV = NH_*HD_ + 2*NKV_*HD_;   // 1536 fused qkv output cols

typedef __attribute__((ext_vector_type(16))) __bf16 v16bf;
typedef __attribute__((ext_vector_type(8)))  __bf16 v8bf;
typedef __attribute__((ext_vector_type(8)))  float  v8f;
typedef __attribute__((ext_vector_type(4)))  int    v4i;

#if defined(__has_builtin)
#if __has_builtin(__builtin_amdgcn_global_load_async_to_lds_b128)
#define ATH_HAVE_ASYNC_LDS 1
#endif
#endif

// ---------------------------------------------------------------------------
// WMMA operand loaders.  Layouts per CDNA5 ISA §7.12.2 (wave32):
//  A (16x32 bf16): lane L holds row M=L&15; halves h -> K = h + 8*(L>=16) + 8*(h>=8)
//  B (32x16 bf16): lane L holds col N=L&15; halves h -> K = h + 8*(L>=16) + 8*(h>=8)
//  C/D (16x16 f32): lane L holds col N=L&15; VGPR r -> M = r + 8*(L>=16)
// ---------------------------------------------------------------------------
__device__ __forceinline__ v16bf load_a16x32(const __bf16* base, int ld, int lane) {
  const int m  = lane & 15;
  const int k0 = (lane >> 4) << 3;                 // 0 or 8
  const __bf16* p = base + m * ld + k0;
  v8bf x0 = *(const v8bf*)(p);                     // K = k0 .. k0+7
  v8bf x1 = *(const v8bf*)(p + 16);                // K = k0+16 .. k0+23
  v16bf r;
#pragma unroll
  for (int i = 0; i < 8; ++i) { r[i] = x0[i]; r[i + 8] = x1[i]; }
  return r;
}

// Loads B operand from "B-transposed" storage: src is [N][K] row-major, so each
// lane's 16 K-values come as two 16B runs (K = k0..k0+7, k0+16..k0+23).
__device__ __forceinline__ v16bf load_b32x16T(const __bf16* base, int ld, int lane) {
  const int n  = lane & 15;
  const int k0 = (lane >> 4) << 3;                 // 0 or 8
  const __bf16* p = base + n * ld + k0;
  v8bf x0 = *(const v8bf*)(p);
  v8bf x1 = *(const v8bf*)(p + 16);
  v16bf r;
#pragma unroll
  for (int i = 0; i < 8; ++i) { r[i] = x0[i]; r[i + 8] = x1[i]; }
  return r;
}

__device__ __forceinline__ v8f wmma_bf16(v16bf a, v16bf b, v8f c) {
  return __builtin_amdgcn_wmma_f32_16x16x32_bf16(
      /*neg_a=*/false, a, /*neg_b=*/false, b,
      /*c_mod=*/(short)0, c, /*reuse_a=*/false, /*reuse_b=*/false);
}

// 16-byte global -> LDS copy.  Async (ASYNCcnt-tracked, no VGPR transit) when
// the builtin exists; plain load + ds_store otherwise.  The builtin signature
// (leaked by clang's diagnostic) takes int4 pointers: AS1 src, AS3 dst.
__device__ __forceinline__ void copy16_g2l(const __bf16* g, __bf16* l) {
#ifdef ATH_HAVE_ASYNC_LDS
  typedef __attribute__((address_space(1))) v4i gv4i;
  typedef __attribute__((address_space(3))) v4i lv4i;
  __builtin_amdgcn_global_load_async_to_lds_b128(
      (gv4i*)const_cast<__bf16*>(g), (lv4i*)l, /*offset=*/0, /*cpol=*/0);
#else
  *(v8bf*)l = *(const v8bf*)g;
#endif
}

__device__ __forceinline__ void wait_async_copies() {
#ifdef ATH_HAVE_ASYNC_LDS
#if __has_builtin(__builtin_amdgcn_s_wait_asynccnt)
  __builtin_amdgcn_s_wait_asynccnt(0);
#else
  asm volatile("s_wait_asynccnt 0" ::: "memory");
#endif
#endif
}

// ---------------------------------------------------------------------------
// LDS-staged WMMA GEMM: C[M][N] f32 = A[M][K] bf16 (row-major) x Bt[N][K] bf16.
// 256 threads = 8 waves per block, block tile 128x128, K-step 32.
// Per k-step: 512+512 16B chunks of A/B are async-copied into LDS (8KB+8KB),
// then wave w (2x4 grid) computes a 64x32 sub-tile: 4 A tiles x 2 B tiles
// -> 8 WMMA.  64 WMMA/block/k-step == the tile's exact math.  This cuts L2
// re-streaming ~4x vs one-wave 64x64 tiles and keeps transit out of VGPRs.
// ---------------------------------------------------------------------------
__global__ __launch_bounds__(256)
void wmma_gemm_bf16_lds(const __bf16* __restrict__ A, const __bf16* __restrict__ Bt,
                        float* __restrict__ C, int M, int N, int K) {
  constexpr int BM = 128, BN = 128, BK = 32;
  __shared__ __align__(16) __bf16 lA[BM * BK];   // 8KB, [m][k]
  __shared__ __align__(16) __bf16 lB[BN * BK];   // 8KB, [n][k]

  const int tid  = threadIdx.x;
  const int lane = tid & 31;
  const int w    = tid >> 5;           // wave 0..7
  const int wm   = (w >> 2) * 64;      // wave M offset within block tile
  const int wn   = (w & 3) * 32;       // wave N offset within block tile
  const int m0   = blockIdx.x * BM;
  const int n0   = blockIdx.y * BN;

  v8f vz = {0, 0, 0, 0, 0, 0, 0, 0};
  v8f acc[4][2];
#pragma unroll
  for (int i = 0; i < 4; ++i)
#pragma unroll
    for (int j = 0; j < 2; ++j) acc[i][j] = vz;

  for (int k = 0; k < K; k += BK) {
    // Stage A and B tiles: 128 rows x 32 halves = 512 chunks of 16B each.
#pragma unroll
    for (int c = 0; c < 2; ++c) {
      const int chunk = tid + c * 256;       // 0..511
      const int row   = chunk >> 2;
      const int col   = (chunk & 3) * 8;     // halves
      copy16_g2l(A  + (size_t)(m0 + row) * K + k + col, lA + row * BK + col);
      copy16_g2l(Bt + (size_t)(n0 + row) * K + k + col, lB + row * BK + col);
    }
    wait_async_copies();
    __syncthreads();

    v16bf a[4];
#pragma unroll
    for (int mt = 0; mt < 4; ++mt)
      a[mt] = load_a16x32(lA + (wm + mt * 16) * BK, BK, lane);
#pragma unroll
    for (int nt = 0; nt < 2; ++nt) {
      v16bf b = load_b32x16T(lB + (wn + nt * 16) * BK, BK, lane);
#pragma unroll
      for (int mt = 0; mt < 4; ++mt) acc[mt][nt] = wmma_bf16(a[mt], b, acc[mt][nt]);
    }
    __syncthreads();                         // before next k-step overwrites LDS
  }

  const int cn  = lane & 15;
  const int cm0 = (lane >> 4) << 3;
#pragma unroll
  for (int mt = 0; mt < 4; ++mt)
#pragma unroll
    for (int nt = 0; nt < 2; ++nt)
#pragma unroll
      for (int r = 0; r < 8; ++r)
        C[(size_t)(m0 + wm + mt * 16 + cm0 + r) * N + n0 + wn + nt * 16 + cn] =
            acc[mt][nt][r];
}

// ---------------------------------------------------------------------------
// Converters
// ---------------------------------------------------------------------------
__global__ void cvt_f32_bf16(const float* __restrict__ s, __bf16* __restrict__ d, int n) {
  int i = blockIdx.x * blockDim.x + threadIdx.x;
  if (i < n) d[i] = (__bf16)s[i];
}

// w: [1024][N] f32  ->  wt: [N][1024] bf16  (transpose so GEMM B-operand lanes
// read contiguous K).
__global__ void cvt_wT(const float* __restrict__ w, __bf16* __restrict__ wt, int N) {
  int i = blockIdx.x * blockDim.x + threadIdx.x;
  if (i >= N * 1024) return;
  int k = i & 1023;
  int n = i >> 10;
  wt[(size_t)n * 1024 + k] = (__bf16)w[(size_t)k * N + n];
}

// ---------------------------------------------------------------------------
// Epilogue for q/k: RoPE + RMSNorm, f32 -> bf16, q laid out [b][h][t][d],
// k laid out [b][hk][t][d] (rows become WMMA B-columns for Q K^T).
// ---------------------------------------------------------------------------
__global__ void epilogue_qk(const float* __restrict__ qkv, const float* __restrict__ cosb,
                            const float* __restrict__ sinb, __bf16* __restrict__ qb,
                            __bf16* __restrict__ kb) {
  int tid = blockIdx.x * blockDim.x + threadIdx.x;
  const int total = B_ * T_ * (NH_ + NKV_);
  if (tid >= total) return;
  const int hs = tid % (NH_ + NKV_);
  int bt = tid / (NH_ + NKV_);
  const int t = bt % T_;
  const int b = bt / T_;

  const float* src;
  __bf16* dst;
  if (hs < NH_) {
    src = qkv + (size_t)(b * T_ + t) * NQKV + hs * HD_;
    dst = qb + ((size_t)(b * NH_ + hs) * T_ + t) * HD_;
  } else {
    int hk = hs - NH_;
    src = qkv + (size_t)(b * T_ + t) * NQKV + NH_ * HD_ + hk * HD_;
    dst = kb + ((size_t)(b * NKV_ + hk) * T_ + t) * HD_;
  }
  const float* cp = cosb + (size_t)t * (HD_ / 2);
  const float* sp = sinb + (size_t)t * (HD_ / 2);

  float v[HD_];
  float ss = 0.f;
#pragma unroll
  for (int i = 0; i < HD_ / 2; ++i) {
    float x1 = src[i], x2 = src[i + HD_ / 2];
    float c = cp[i], s = sp[i];
    float a = x1 * c + x2 * s;
    float d = -x1 * s + x2 * c;
    v[i] = a;
    v[i + HD_ / 2] = d;
    ss += a * a + d * d;
  }
  float r = rsqrtf(ss * (1.f / HD_) + 1.1920929e-7f);
#pragma unroll
  for (int i = 0; i < HD_; ++i) dst[i] = (__bf16)(v[i] * r);
}

// ---------------------------------------------------------------------------
// Epilogue for v: v += 2*sigmoid(x[:,:32]@Wgate) * ve, stored TRANSPOSED as
// vt[b][hk][d][t] so the PV WMMA B-operand reads contiguous keys per lane.
// ---------------------------------------------------------------------------
__global__ void epilogue_v(const float* __restrict__ qkv, const float* __restrict__ x,
                           const float* __restrict__ ve, const float* __restrict__ wg,
                           __bf16* __restrict__ vt) {
  int tid = blockIdx.x * blockDim.x + threadIdx.x;
  if (tid >= B_ * NKV_ * HD_ * T_) return;
  const int t  = tid & (T_ - 1);
  const int d  = (tid >> 11) & (HD_ - 1);
  const int hk = (tid >> 17) & (NKV_ - 1);
  const int b  = tid >> 19;

  const float* xp = x + (size_t)(b * T_ + t) * C_;
  float acc = 0.f;
#pragma unroll
  for (int i = 0; i < VEC_; ++i) acc += xp[i] * wg[i * NKV_ + hk];
  float g = 2.f / (1.f + __expf(-acc));

  float val = qkv[(size_t)(b * T_ + t) * NQKV + (NH_ + NKV_) * HD_ + hk * HD_ + d]
            + g * ve[(size_t)(b * T_ + t) * (NKV_ * HD_) + hk * HD_ + d];
  vt[((size_t)(b * NKV_ + hk) * HD_ + d) * T_ + t] = (__bf16)val;
}

// ---------------------------------------------------------------------------
// Flash attention, sliding causal window.  One wave per (b, h, 16-row q tile).
// Per 32-key step: 4 WMMA for S = Q K^T (2 subtiles x 2 d-chunks), LDS
// transpose of P (C-layout -> A-layout), 4 WMMA for O += P V.
// Row softmax stats via __shfl_xor width-16 (C-layout spreads a row over the
// 16 lanes of a half-wave).  Single-wave workgroup => barriers are NOP-cheap.
// ---------------------------------------------------------------------------
__global__ __launch_bounds__(32)
void flash_attn_swa(const __bf16* __restrict__ qb, const __bf16* __restrict__ kb,
                    const __bf16* __restrict__ vt, __bf16* __restrict__ yb,
                    const int* __restrict__ winp) {
  __shared__ __align__(16) __bf16 ldsP[16 * 32];

  const int lane = threadIdx.x & 31;
  const int hiw  = lane >> 4;       // 0/1 half-wave
  const int lr   = lane & 15;

  int blk = blockIdx.x;
  const int qt = blk & (T_ / 16 - 1); blk >>= 7;   // T_/16 = 128
  const int h  = blk & (NH_ - 1);     blk >>= 4;
  const int b  = blk;
  const int hk = h / (NH_ / NKV_);
  const int q0 = qt * 16;
  const int W  = *winp;

  const __bf16* qptr = qb + ((size_t)(b * NH_ + h) * T_ + q0) * HD_;
  const __bf16* kptr = kb + (size_t)(b * NKV_ + hk) * T_ * HD_;
  const __bf16* vptr = vt + (size_t)(b * NKV_ + hk) * HD_ * T_;

  const v16bf qa0 = load_a16x32(qptr, HD_, lane);       // d = 0..31
  const v16bf qa1 = load_a16x32(qptr + 32, HD_, lane);  // d = 32..63

  v8f vz = {0, 0, 0, 0, 0, 0, 0, 0};
  float mrow[8], lrow[8];
  v8f o[4];
#pragma unroll
  for (int r = 0; r < 8; ++r) { mrow[r] = -3.0e38f; lrow[r] = 0.f; }
#pragma unroll
  for (int nt = 0; nt < 4; ++nt) o[nt] = vz;

  int jlo = q0 - W;
  if (jlo < 0) jlo = 0;
  jlo &= ~31;
  const int jhi = q0 + 15;

  for (int j0 = jlo; j0 <= jhi; j0 += 32) {
    __builtin_prefetch(kptr + (size_t)(j0 + 32) * HD_, 0, 1);  // global_prefetch_b8

    // S = Q K^T  (two 16-key subtiles, accumulate over both 32-wide d chunks)
    v8f s[2];
#pragma unroll
    for (int sub = 0; sub < 2; ++sub) {
      const int jb = j0 + sub * 16;
      v8f sc = vz;
      v16bf bk0 = load_b32x16T(kptr + (size_t)jb * HD_, HD_, lane);
      sc = wmma_bf16(qa0, bk0, sc);
      v16bf bk1 = load_b32x16T(kptr + (size_t)jb * HD_ + 32, HD_, lane);
      sc = wmma_bf16(qa1, bk1, sc);
      s[sub] = sc;
    }

    // Mask + online softmax update (flash).  Element (r): row = q0+8*hiw+r,
    // col = j0+16*sub+lr (C/D layout).
#pragma unroll
    for (int r = 0; r < 8; ++r) {
      const int row = q0 + hiw * 8 + r;
#pragma unroll
      for (int sub = 0; sub < 2; ++sub) {
        const int col = j0 + sub * 16 + lr;
        const bool ok = (col <= row) && (col >= row - W);
        s[sub][r] = ok ? s[sub][r] * 0.125f : -3.0e38f;   // 1/sqrt(64)
      }
      float mv = fmaxf(s[0][r], s[1][r]);
      mv = fmaxf(mv, __shfl_xor(mv, 1, 16));
      mv = fmaxf(mv, __shfl_xor(mv, 2, 16));
      mv = fmaxf(mv, __shfl_xor(mv, 4, 16));
      mv = fmaxf(mv, __shfl_xor(mv, 8, 16));
      const float mn   = fmaxf(mrow[r], mv);
      const float resc = __expf(mrow[r] - mn);
      mrow[r] = mn;
      const float p0 = __expf(s[0][r] - mn);
      const float p1 = __expf(s[1][r] - mn);
      float ps = p0 + p1;
      ps += __shfl_xor(ps, 1, 16);
      ps += __shfl_xor(ps, 2, 16);
      ps += __shfl_xor(ps, 4, 16);
      ps += __shfl_xor(ps, 8, 16);
      lrow[r] = lrow[r] * resc + ps;
#pragma unroll
      for (int nt = 0; nt < 4; ++nt) o[nt][r] *= resc;
      // C-layout -> LDS row-major P[16][32]
      ldsP[(hiw * 8 + r) * 32 + lr]      = (__bf16)p0;
      ldsP[(hiw * 8 + r) * 32 + 16 + lr] = (__bf16)p1;
    }
    __syncthreads();                       // single-wave WG: barrier==NOP + DS fence
    const v16bf pa = load_a16x32(ldsP, 32, lane);
    __syncthreads();

    // O += P V   (vt is [d][t] so each lane reads contiguous keys)
#pragma unroll
    for (int nt = 0; nt < 4; ++nt) {
      v16bf bv = load_b32x16T(vptr + (size_t)(nt * 16) * T_ + j0, T_, lane);
      o[nt] = wmma_bf16(pa, bv, o[nt]);
    }
  }

  // y[b][t][h*64 + d] bf16 (feeds the output-projection GEMM)
  __bf16* yp = yb + ((size_t)b * T_ + q0) * C_ + h * HD_;
#pragma unroll
  for (int nt = 0; nt < 4; ++nt)
#pragma unroll
    for (int r = 0; r < 8; ++r)
      yp[(size_t)(hiw * 8 + r) * C_ + nt * 16 + lr] = (__bf16)(o[nt][r] / lrow[r]);
}

// ---------------------------------------------------------------------------
// Host-side orchestration
// ---------------------------------------------------------------------------
extern "C" void kernel_launch(void* const* d_in, const int* in_sizes, int n_in,
                              void* d_out, int out_size, void* d_ws, size_t ws_size,
                              hipStream_t stream) {
  const float* x    = (const float*)d_in[0];
  const float* ve   = (const float*)d_in[1];
  const float* cosb = (const float*)d_in[2];
  const float* sinb = (const float*)d_in[3];
  const float* Wq   = (const float*)d_in[4];
  const float* Wk   = (const float*)d_in[5];
  const float* Wv   = (const float*)d_in[6];
  const float* Wp   = (const float*)d_in[7];
  const float* Wg   = (const float*)d_in[8];
  const int*   win  = (const int*)d_in[9];
  float* out = (float*)d_out;

  char* ws = (char*)d_ws;
  size_t off = 0;
  auto carve = [&](size_t bytes) -> char* {
    char* p = ws + off;
    off += (bytes + 255) & ~(size_t)255;
    return p;
  };
  __bf16* xb    = (__bf16*)carve((size_t)B_ * T_ * C_ * 2);          // x bf16
  __bf16* wqkvt = (__bf16*)carve((size_t)NQKV * C_ * 2);             // [Wq|Wk|Wv]^T bf16
  __bf16* wpt   = (__bf16*)carve((size_t)C_ * C_ * 2);               // Wproj^T bf16
  float*  qkvr  = (float*) carve((size_t)B_ * T_ * NQKV * 4);        // fused qkv f32
  __bf16* qb    = (__bf16*)carve((size_t)B_ * NH_ * T_ * HD_ * 2);   // q [b][h][t][d]
  __bf16* kb    = (__bf16*)carve((size_t)B_ * NKV_ * T_ * HD_ * 2);  // k [b][hk][t][d]
  __bf16* vt    = (__bf16*)carve((size_t)B_ * NKV_ * HD_ * T_ * 2);  // v [b][hk][d][t]
  __bf16* yb    = (__bf16*)carve((size_t)B_ * T_ * C_ * 2);          // attn out bf16

  const int nx = B_ * T_ * C_;
  cvt_f32_bf16<<<(nx + 255) / 256, 256, 0, stream>>>(x, xb, nx);
  cvt_wT<<<(1024 * 1024 + 255) / 256, 256, 0, stream>>>(Wq, wqkvt, NH_ * HD_);
  cvt_wT<<<(1024 * 256 + 255) / 256, 256, 0, stream>>>(
      Wk, wqkvt + (size_t)NH_ * HD_ * C_, NKV_ * HD_);
  cvt_wT<<<(1024 * 256 + 255) / 256, 256, 0, stream>>>(
      Wv, wqkvt + (size_t)(NH_ + NKV_) * HD_ * C_, NKV_ * HD_);
  cvt_wT<<<(1024 * 1024 + 255) / 256, 256, 0, stream>>>(Wp, wpt, C_);

  // Fused QKV projection: (4096 x 1024) x (1024 x 1536)
  wmma_gemm_bf16_lds<<<dim3(B_ * T_ / 128, NQKV / 128), 256, 0, stream>>>(
      xb, wqkvt, qkvr, B_ * T_, NQKV, C_);

  const int nqk = B_ * T_ * (NH_ + NKV_);
  epilogue_qk<<<(nqk + 255) / 256, 256, 0, stream>>>(qkvr, cosb, sinb, qb, kb);
  const int nv = B_ * NKV_ * HD_ * T_;
  epilogue_v<<<(nv + 255) / 256, 256, 0, stream>>>(qkvr, x, ve, Wg, vt);

  flash_attn_swa<<<B_ * NH_ * (T_ / 16), 32, 0, stream>>>(qb, kb, vt, yb, win);

  // Output projection: (4096 x 1024) x (1024 x 1024) -> d_out f32
  wmma_gemm_bf16_lds<<<dim3(B_ * T_ / 128, C_ / 128), 256, 0, stream>>>(
      yb, wpt, out, B_ * T_, C_, C_);
}
